// DAG_GNN_13194139533784
// MI455X (gfx1250) — matrix operand order repaired and checked by reading, not compiled, not run on hardware
//
#include <hip/hip_runtime.h>
#include <hip/hip_bf16.h>
#include <cstdint>

typedef _Float16 half_t;
typedef __attribute__((ext_vector_type(16))) _Float16 v16h;
typedef __attribute__((ext_vector_type(8)))  float    v8f;

#define B_    512
#define N_    128
#define HD_   256
#define HD3_  768
#define VT_   9
#define NVAR_ 3
#define Z_    64
#define TOPO_ 12

#define WMMA_F16(a, b, c) \
  __builtin_amdgcn_wmma_f32_16x16x32_f16(false, (a), false, (b), (short)0, (c), false, false)

__device__ __forceinline__ float sigm(float x) { return 1.0f / (1.0f + __expf(-x)); }

// ---- CDNA5 wave32 WMMA fragment index helpers (CDNA5 ISA §7.12.2) ----
// A-matrix 16x32 f16: lanes 0-15 row M=lane, halves[0..7]=K0..7, halves[8..15]=K16..23
//                     lanes 16-31 row M=lane-16, halves = K8..15 / K24..31
__device__ __forceinline__ size_t afrag_idx(int m, int k, int KT) {
  int mt = m >> 4, rb = m & 15;
  int kt = k >> 5, ko = k & 31;
  int hi = ko >> 3, lo = ko & 7;
  int lane = rb + ((hi & 1) << 4);
  int s    = ((hi >> 1) << 3) + lo;
  return ((((size_t)mt * KT + kt) * 32 + lane) << 4) + s;
}

// ---- pack f32 weight [srcK x srcN] into B-fragment layout (K padded to Kpad) ----
// B-matrix 32x16 f16: lane = col%16 + ((k%32)/8 odd ? 16:0), half slot s = ((k%32)/16)*8 + k%8
__global__ void pack_b_kernel(const float* __restrict__ src, half_t* __restrict__ dst,
                              int srcK, int srcN, int Kpad, int KT, int n0) {
  int idx = blockIdx.x * 256 + threadIdx.x;
  int total = Kpad * srcN;
  if (idx >= total) return;
  int k = idx / srcN, n = idx % srcN;
  float v = (k < srcK) ? src[(size_t)k * srcN + n] : 0.0f;
  int col = n + n0;
  int nt = col >> 4;
  int ko = k & 31;
  int lane = (col & 15) + (((ko >> 3) & 1) << 4);
  int s = ((ko >> 4) << 3) + (ko & 7);
  dst[((((size_t)nt * KT + (k >> 5)) * 32 + lane) << 4) + s] = (half_t)v;
}

__global__ void pack_a_kernel(const float* __restrict__ src, half_t* __restrict__ dst,
                              int K, int KT, int total) {
  int idx = blockIdx.x * 256 + threadIdx.x;
  if (idx >= total) return;
  int m = idx / K, k = idx % K;
  dst[afrag_idx(m, k, KT)] = (half_t)src[idx];
}

// ---- per-step: msg gather over G + pack [x_v | msg] into A-fragments ----
__global__ void msg_pack_kernel(const float* __restrict__ G, const float* __restrict__ adj,
                                const float* __restrict__ Hs, const float* __restrict__ feats,
                                float* __restrict__ msg, half_t* __restrict__ Acat,
                                int v, int reverse, int layer0) {
  __shared__ float sA[N_];
  int b = blockIdx.x, t = threadIdx.x;
  if (t < N_)
    sA[t] = reverse ? adj[((size_t)b * N_ + t) * N_ + v]
                    : adj[((size_t)b * N_ + v) * N_ + t];
  __syncthreads();
  int n0 = reverse ? v + 1 : 0;
  int n1 = reverse ? N_ : v;
  float acc = 0.0f;
  for (int n = n0; n < n1; ++n) {
    float a = sA[n];                 // block-uniform -> scalar-branch skip of zeros
    if (a != 0.0f) acc += a * G[(((size_t)b * N_ + n) << 8) + t];
  }
  msg[((size_t)b << 8) + t] = acc;
  float xv;
  if (layer0) xv = (t < VT_) ? feats[((size_t)b * N_ + v) * VT_ + t] : 0.0f;
  else        xv = Hs[(((size_t)b * N_ + v) << 8) + t];
  Acat[afrag_idx(b, t,       32)] = (half_t)xv;
  Acat[afrag_idx(b, 256 + t, 32)] = (half_t)acc;
}

// ---- dual GEMM (z=0: x@Wx, z=1: msg@Wh), N=768, 4 N-tiles per wave ----
__global__ __launch_bounds__(32) void gemm_xh_kernel(
    const half_t* __restrict__ Ap0, const half_t* __restrict__ Ap1,
    const half_t* __restrict__ Bp0, const half_t* __restrict__ Bp1,
    float* __restrict__ C0, float* __restrict__ C1,
    int ktCount, int aKT) {
  int mt = blockIdx.x, ng = blockIdx.y, z = blockIdx.z;
  int lane = threadIdx.x;
  const half_t* Ap = z ? Ap1 : Ap0;
  const half_t* Bp = z ? Bp1 : Bp0;
  float* C = z ? C1 : C0;
  int nt0 = ng * 4;
  const v16h* a  = (const v16h*)(Ap + (((size_t)mt * aKT) * 32 + lane) * 16);
  const v16h* b0 = (const v16h*)(Bp + (((size_t)nt0 * ktCount) * 32 + lane) * 16);
  size_t bs = (size_t)ktCount * 32;     // v16h per N-tile
  v8f acc[4] = {{}, {}, {}, {}};
  for (int kt = 0; kt < ktCount; ++kt) {
    v16h af  = a[(size_t)kt * 32];
    v16h bf0 = b0[(size_t)kt * 32];
    v16h bf1 = b0[bs     + (size_t)kt * 32];
    v16h bf2 = b0[2 * bs + (size_t)kt * 32];
    v16h bf3 = b0[3 * bs + (size_t)kt * 32];
    acc[0] = WMMA_F16(af, bf0, acc[0]);
    acc[1] = WMMA_F16(af, bf1, acc[1]);
    acc[2] = WMMA_F16(af, bf2, acc[2]);
    acc[3] = WMMA_F16(af, bf3, acc[3]);
  }
  int row0 = mt * 16 + ((lane >> 4) << 3);
#pragma unroll
  for (int j = 0; j < 4; ++j) {
    int col = (nt0 + j) * 16 + (lane & 15);
#pragma unroll
    for (int i = 0; i < 8; ++i)
      C[(size_t)(row0 + i) * HD3_ + col] = acc[j][i];
  }
}

// ---- gate GEMM h_new @ [Wg | Wm], fused sigmoid(.)+bg * (.) epilogue via LDS ----
// wave0: 4 N-tiles of Wg columns, wave1: matching 4 N-tiles of Wm columns
__global__ __launch_bounds__(64) void gemm_gate_kernel(
    const half_t* __restrict__ Ap, const half_t* __restrict__ Bp,
    const float* __restrict__ bg, float* __restrict__ G, int v) {
  __shared__ float xch[32][4][8];
  int mt = blockIdx.x, pr = blockIdx.y;
  int wave = threadIdx.x >> 5, lane = threadIdx.x & 31;
  int nt0 = pr * 4 + wave * 16;
  const v16h* a  = (const v16h*)(Ap + (((size_t)mt * 16) * 32 + lane) * 16);
  const v16h* b0 = (const v16h*)(Bp + (((size_t)nt0 * 16) * 32 + lane) * 16);
  const size_t bs = 16 * 32;
  v8f acc[4] = {{}, {}, {}, {}};
  for (int kt = 0; kt < 16; ++kt) {
    v16h af  = a[(size_t)kt * 32];
    v16h bf0 = b0[(size_t)kt * 32];
    v16h bf1 = b0[bs     + (size_t)kt * 32];
    v16h bf2 = b0[2 * bs + (size_t)kt * 32];
    v16h bf3 = b0[3 * bs + (size_t)kt * 32];
    acc[0] = WMMA_F16(af, bf0, acc[0]);
    acc[1] = WMMA_F16(af, bf1, acc[1]);
    acc[2] = WMMA_F16(af, bf2, acc[2]);
    acc[3] = WMMA_F16(af, bf3, acc[3]);
  }
  if (wave == 1) {
#pragma unroll
    for (int j = 0; j < 4; ++j)
#pragma unroll
      for (int i = 0; i < 8; ++i) xch[lane][j][i] = acc[j][i];
  }
  __syncthreads();
  if (wave == 0) {
    int row0 = mt * 16 + ((lane >> 4) << 3);
#pragma unroll
    for (int j = 0; j < 4; ++j) {
      int col = (pr * 4 + j) * 16 + (lane & 15);
      float bgv = bg[col];
#pragma unroll
      for (int i = 0; i < 8; ++i) {
        float g = sigm(acc[j][i] + bgv) * xch[lane][j][i];
        G[(((size_t)(row0 + i) * N_ + v) << 8) + col] = g;
      }
    }
  }
}

// ---- GRU nonlinearity; writes f32 state + A-fragment pack of h_new ----
__global__ void gru_kernel(const float* __restrict__ xo, const float* __restrict__ ho,
                           const float* __restrict__ hprev,
                           const float* __restrict__ bx, const float* __restrict__ bh,
                           float* __restrict__ outF, int outStride,
                           half_t* __restrict__ packOut, int packKT) {
  int b = blockIdx.x, t = threadIdx.x;
  size_t o = (size_t)b * HD3_;
  float xr = xo[o + t]       + bx[t];
  float xz = xo[o + 256 + t] + bx[256 + t];
  float xn = xo[o + 512 + t] + bx[512 + t];
  float hr = ho[o + t]       + bh[t];
  float hz = ho[o + 256 + t] + bh[256 + t];
  float hn = ho[o + 512 + t] + bh[512 + t];
  float r = sigm(xr + hr);
  float z = sigm(xz + hz);
  float n = tanhf(xn + r * hn);
  float hp = hprev[((size_t)b << 8) + t];
  float h  = (1.0f - z) * n + z * hp;
  outF[(size_t)b * outStride + t] = h;
  packOut[afrag_idx(b, t, packKT)] = (half_t)h;
}

// ---- variable readout gather -> A-fragment pack (M = B*NVAR) ----
__global__ void var_gather_kernel(const float* __restrict__ Hs,
                                  const int* __restrict__ var_pos,
                                  half_t* __restrict__ dst) {
  int m = blockIdx.x, t = threadIdx.x;  // m = b*NVAR + i
  int b = m / NVAR_, i = m % NVAR_;
  int p = var_pos[b * NVAR_ + i];
  float v = Hs[(((size_t)b * N_ + p) << 8) + t];
  dst[afrag_idx(m, t, 16)] = (half_t)v;
}

// ---- generic GEMM with bias epilogue (head mu/sg projection), 4 N-tiles/wave ----
__global__ __launch_bounds__(32) void gemm_bias_kernel(
    const half_t* __restrict__ Ap, const half_t* __restrict__ Bp,
    float* __restrict__ C, const float* __restrict__ bias,
    int ktCount, int aKT, int ldc) {
  int mt = blockIdx.x, ng = blockIdx.y, lane = threadIdx.x;
  int nt0 = ng * 4;
  const v16h* a  = (const v16h*)(Ap + (((size_t)mt * aKT) * 32 + lane) * 16);
  const v16h* b0 = (const v16h*)(Bp + (((size_t)nt0 * ktCount) * 32 + lane) * 16);
  size_t bs = (size_t)ktCount * 32;
  v8f acc[4] = {{}, {}, {}, {}};
  for (int kt = 0; kt < ktCount; ++kt) {
    v16h af  = a[(size_t)kt * 32];
    v16h bf0 = b0[(size_t)kt * 32];
    v16h bf1 = b0[bs     + (size_t)kt * 32];
    v16h bf2 = b0[2 * bs + (size_t)kt * 32];
    v16h bf3 = b0[3 * bs + (size_t)kt * 32];
    acc[0] = WMMA_F16(af, bf0, acc[0]);
    acc[1] = WMMA_F16(af, bf1, acc[1]);
    acc[2] = WMMA_F16(af, bf2, acc[2]);
    acc[3] = WMMA_F16(af, bf3, acc[3]);
  }
  int row0 = mt * 16 + ((lane >> 4) << 3);
#pragma unroll
  for (int j = 0; j < 4; ++j) {
    int col = (nt0 + j) * 16 + (lane & 15);
    float bv = bias[col];
#pragma unroll
    for (int i = 0; i < 8; ++i)
      C[(size_t)(row0 + i) * ldc + col] = acc[j][i] + bv;
  }
}

__global__ void concat_bias_kernel(const float* __restrict__ a, const float* __restrict__ b,
                                   float* __restrict__ out) {
  int t = threadIdx.x;
  out[t] = (t < Z_) ? a[t] : b[t - Z_];
}

// ---- final tiny head: [mu|topo]@Wmt+bmt , [sg|topo]@Wst+bst (K=76, scalar f32) ----
__global__ void head_final_kernel(const float* __restrict__ musg, const float* __restrict__ topo,
                                  const float* __restrict__ Wmt, const float* __restrict__ bmt,
                                  const float* __restrict__ Wst, const float* __restrict__ bst,
                                  float* __restrict__ out) {
  int b = blockIdx.x, j = threadIdx.x;
  int hi = (j >= Z_);
  int jj = j & (Z_ - 1);
  const float* W = hi ? Wst : Wmt;
  float acc = hi ? bst[jj] : bmt[jj];
  int base = hi ? Z_ : 0;
  for (int t2 = 0; t2 < Z_; ++t2)
    acc += musg[(size_t)b * 2 * Z_ + base + t2] * W[t2 * Z_ + jj];
  for (int t2 = 0; t2 < TOPO_; ++t2)
    acc += topo[(size_t)b * TOPO_ + t2] * W[(Z_ + t2) * Z_ + jj];
  out[(size_t)b * 2 * Z_ + j] = acc;
}

extern "C" void kernel_launch(void* const* d_in, const int* in_sizes, int n_in,
                              void* d_out, int out_size, void* d_ws, size_t ws_size,
                              hipStream_t stream) {
  (void)in_sizes; (void)n_in; (void)out_size; (void)ws_size;
  const float* feats = (const float*)d_in[0];
  const float* adj   = (const float*)d_in[1];
  const float* topo  = (const float*)d_in[2];
  const int*   vpos  = (const int*)d_in[3];
  const float* Wx0f  = (const float*)d_in[4];
  const float* Wh0f  = (const float*)d_in[5];
  const float* bx0f  = (const float*)d_in[6];
  const float* bh0f  = (const float*)d_in[7];
  const float* Wxf   = (const float*)d_in[8];
  const float* Whf   = (const float*)d_in[9];
  const float* bxf   = (const float*)d_in[10];
  const float* bhf   = (const float*)d_in[11];
  const float* Wxb   = (const float*)d_in[12];
  const float* Whb   = (const float*)d_in[13];
  const float* bxb   = (const float*)d_in[14];
  const float* bhb   = (const float*)d_in[15];
  const float* Wg    = (const float*)d_in[16];
  const float* bg    = (const float*)d_in[17];
  const float* Wm    = (const float*)d_in[18];
  const float* Wxv   = (const float*)d_in[19];
  const float* Whv   = (const float*)d_in[20];
  const float* bxv   = (const float*)d_in[21];
  const float* bhv   = (const float*)d_in[22];
  const float* Wmu   = (const float*)d_in[23];
  const float* bmu   = (const float*)d_in[24];
  const float* Wsg   = (const float*)d_in[25];
  const float* bsg   = (const float*)d_in[26];
  const float* Wmt   = (const float*)d_in[27];
  const float* bmt   = (const float*)d_in[28];
  const float* Wst   = (const float*)d_in[29];
  const float* bst   = (const float*)d_in[30];

  // bump allocator on workspace (256B aligned)
  char* wbase = (char*)d_ws; size_t off = 0;
  auto alloc = [&](size_t bytes) -> void* {
    void* p = wbase + off;
    off = (off + bytes + 255) & ~(size_t)255;
    return p;
  };

  float* Hs    = (float*)alloc((size_t)B_ * N_ * HD_ * 4);    // 64 MB (L2 resident)
  float* G     = (float*)alloc((size_t)B_ * N_ * HD_ * 4);    // 64 MB (L2 resident)
  float* msg   = (float*)alloc((size_t)B_ * HD_ * 4);
  float* xout  = (float*)alloc((size_t)(B_ * NVAR_) * HD3_ * 4);
  float* hout  = (float*)alloc((size_t)(B_ * NVAR_) * HD3_ * 4);
  float* hv    = (float*)alloc((size_t)(B_ * NVAR_) * HD_ * 4);
  float* musg  = (float*)alloc((size_t)B_ * 2 * Z_ * 4);
  float* bmusg = (float*)alloc(2 * Z_ * 4);

  const size_t W768 = (size_t)HD_ * HD3_;  // elements per 256x768 matrix
  half_t* pWx0   = (half_t*)alloc(W768 * 2);
  half_t* pWh0   = (half_t*)alloc(W768 * 2);
  half_t* pWxf0  = (half_t*)alloc(W768 * 2);
  half_t* pWhf0  = (half_t*)alloc(W768 * 2);
  half_t* pWxf1  = (half_t*)alloc(W768 * 2);
  half_t* pWhf1  = (half_t*)alloc(W768 * 2);
  half_t* pWxb0  = (half_t*)alloc(W768 * 2);
  half_t* pWhb0  = (half_t*)alloc(W768 * 2);
  half_t* pWxb1  = (half_t*)alloc(W768 * 2);
  half_t* pWhb1  = (half_t*)alloc(W768 * 2);
  half_t* pWgm   = (half_t*)alloc((size_t)HD_ * 512 * 2);
  half_t* pWxv   = (half_t*)alloc(W768 * 2);
  half_t* pWhv   = (half_t*)alloc(W768 * 2);
  half_t* pWmusg = (half_t*)alloc((size_t)HD3_ * 128 * 2);
  half_t* Acat   = (half_t*)alloc((size_t)B_ * 512 * 2);
  half_t* ApackH = (half_t*)alloc((size_t)B_ * HD_ * 2);
  half_t* ApackVar[3];
  for (int l = 0; l < 3; ++l) ApackVar[l] = (half_t*)alloc((size_t)(B_ * NVAR_) * HD_ * 2);
  half_t* ApackHv = (half_t*)alloc((size_t)(B_ * NVAR_) * HD_ * 2);
  half_t* ApackHg = (half_t*)alloc((size_t)B_ * HD3_ * 2);

  auto packB = [&](const float* src, half_t* dst, int srcK, int srcN, int Kpad, int KT, int n0) {
    int total = Kpad * srcN;
    pack_b_kernel<<<(total + 255) / 256, 256, 0, stream>>>(src, dst, srcK, srcN, Kpad, KT, n0);
  };
  // weight fragment pre-packing (once per launch; constant across all 640 steps)
  packB(Wx0f,        pWx0,  VT_, HD3_, HD_, 16, 0);   // 9->256 zero-padded K
  packB(Wh0f,        pWh0,  HD_, HD3_, HD_, 16, 0);
  packB(Wxf,         pWxf0, HD_, HD3_, HD_, 16, 0);
  packB(Wxf + W768,  pWxf1, HD_, HD3_, HD_, 16, 0);
  packB(Whf,         pWhf0, HD_, HD3_, HD_, 16, 0);
  packB(Whf + W768,  pWhf1, HD_, HD3_, HD_, 16, 0);
  packB(Wxb,         pWxb0, HD_, HD3_, HD_, 16, 0);
  packB(Wxb + W768,  pWxb1, HD_, HD3_, HD_, 16, 0);
  packB(Whb,         pWhb0, HD_, HD3_, HD_, 16, 0);
  packB(Whb + W768,  pWhb1, HD_, HD3_, HD_, 16, 0);
  packB(Wg,          pWgm,  HD_, HD_,  HD_, 16, 0);
  packB(Wm,          pWgm,  HD_, HD_,  HD_, 16, HD_);
  packB(Wxv,         pWxv,  HD_, HD3_, HD_, 16, 0);
  packB(Whv,         pWhv,  HD_, HD3_, HD_, 16, 0);
  packB(Wmu,         pWmusg, HD3_, Z_, HD3_, 24, 0);
  packB(Wsg,         pWmusg, HD3_, Z_, HD3_, 24, Z_);
  concat_bias_kernel<<<1, 2 * Z_, 0, stream>>>(bmu, bsg, bmusg);

  // zero state each launch (Hs=0, G=gate(0)=0, hv=0)
  (void)hipMemsetAsync(Hs, 0, (size_t)B_ * N_ * HD_ * 4, stream);
  (void)hipMemsetAsync(G,  0, (size_t)B_ * N_ * HD_ * 4, stream);
  (void)hipMemsetAsync(hv, 0, (size_t)(B_ * NVAR_) * HD_ * 4, stream);
  (void)hipMemsetAsync(ApackHv, 0, (size_t)(B_ * NVAR_) * HD_ * 2, stream);

  struct PassDesc {
    int rev, l0, varSlot;
    const half_t *pWx, *pWh;
    const float *bx, *bh;
  };
  PassDesc passes[5] = {
    {0, 1,  0, pWx0,  pWh0,  bx0f,       bh0f},
    {1, 0, -1, pWxb0, pWhb0, bxb,        bhb},
    {0, 0,  1, pWxf0, pWhf0, bxf,        bhf},
    {1, 0, -1, pWxb1, pWhb1, bxb + HD3_, bhb + HD3_},
    {0, 0,  2, pWxf1, pWhf1, bxf + HD3_, bhf + HD3_},
  };

  for (int p = 0; p < 5; ++p) {
    PassDesc& P = passes[p];
    for (int s = 0; s < N_; ++s) {
      int v = P.rev ? (N_ - 1 - s) : s;
      msg_pack_kernel<<<B_, HD_, 0, stream>>>(G, adj, Hs, feats, msg, Acat, v, P.rev, P.l0);
      gemm_xh_kernel<<<dim3(B_ / 16, HD3_ / 64, 2), 32, 0, stream>>>(
          Acat, Acat + 16 * 512, P.pWx, P.pWh, xout, hout, 16, 32);
      gru_kernel<<<B_, HD_, 0, stream>>>(xout, hout, msg, P.bx, P.bh,
                                         Hs + (size_t)v * HD_, N_ * HD_, ApackH, 16);
      gemm_gate_kernel<<<dim3(B_ / 16, 4), 64, 0, stream>>>(ApackH, pWgm, bg, G, v);
    }
    if (P.varSlot >= 0)
      var_gather_kernel<<<B_ * NVAR_, HD_, 0, stream>>>(Hs, vpos, ApackVar[P.varSlot]);
  }

  // GRU over layer axis for the NVAR variables (M = 1536)
  for (int l = 0; l < 3; ++l) {
    gemm_xh_kernel<<<dim3(B_ * NVAR_ / 16, HD3_ / 64, 2), 32, 0, stream>>>(
        ApackVar[l], ApackHv, pWxv, pWhv, xout, hout, 16, 16);
    gru_kernel<<<B_ * NVAR_, HD_, 0, stream>>>(xout, hout, hv, bxv, bhv,
                                               hv, HD_, ApackHv, 16);
  }

  // hv flat == hg [B, NVAR*HD]; repack as M=512,K=768 A-fragments for the head GEMM
  pack_a_kernel<<<(B_ * HD3_ + 255) / 256, 256, 0, stream>>>(hv, ApackHg, HD3_, 24, B_ * HD3_);
  gemm_bias_kernel<<<dim3(B_ / 16, (2 * Z_) / 64), 32, 0, stream>>>(
      ApackHg, pWmusg, musg, bmusg, 24, 24, 2 * Z_);
  head_final_kernel<<<B_, 2 * Z_, 0, stream>>>(musg, topo, Wmt, bmt, Wst, bst, (float*)d_out);
}
